// Model_8727373545523
// MI455X (gfx1250) — matrix operand order repaired and checked by reading, not compiled, not run on hardware
//
#include <hip/hip_runtime.h>
#include <hip/hip_bf16.h>
#include <math.h>

// ---------------------------------------------------------------------------
// Model dims (from reference): B=32, T=256, F=512, M=4, N=508, T2=512
// Rows of the big activation matrix: B*F = 16384. All GEMM dims are multiples
// of the 128x64x32 tile, so no bounds checks are needed on the main path.
// ---------------------------------------------------------------------------

typedef __bf16 bf16_t;
typedef __attribute__((ext_vector_type(16))) __bf16       v16bf;
typedef __attribute__((ext_vector_type(8)))  float        v8f;
typedef __attribute__((ext_vector_type(8)))  unsigned int v8u;
typedef __attribute__((ext_vector_type(4)))  unsigned int v4u;
typedef __attribute__((ext_vector_type(8)))  int          v8i;
typedef __attribute__((ext_vector_type(4)))  int          v4i;

#if __has_builtin(__builtin_amdgcn_tensor_load_to_lds)
#define HAVE_TDM 1
#else
#define HAVE_TDM 0
#endif

#if HAVE_TDM
// ---------------------------------------------------------------------------
// Tensor Data Mover: DMA one 2D tile (tile_w x tile_h elements of 2 bytes,
// row pitch `pitch` elements) from global memory into LDS at byte offset
// lds_off. D# layout per CDNA5 ISA 8.3/8.4 (group0: count/lds/global/type,
// group1: data_size, tensor dims == tile dims, strides; groups 2/3 unused).
// Issue from ONE wave only; complete with s_wait_tensorcnt 0.
// ---------------------------------------------------------------------------
__device__ __forceinline__ void tdm_load_2d(unsigned lds_off, const bf16_t* gptr,
                                            int tile_w, int tile_h, int pitch)
{
    unsigned long long ga = (unsigned long long)(uintptr_t)gptr;
    v4u g0;
    g0[0] = 1u;                                               // count=1, user desc
    g0[1] = lds_off;                                          // lds_addr (bytes)
    g0[2] = (unsigned)ga;                                     // global_addr[31:0]
    g0[3] = ((unsigned)(ga >> 32) & 0x01ffffffu) | 0x80000000u; // addr[56:32] | type=2
    v8i g1;
    g1[0] = 0x00010000;                                       // data_size=1 (2B)
    g1[1] = (int)(((unsigned)tile_w & 0xffffu) << 16);        // tensor_dim0[15:0]
    g1[2] = (int)(((unsigned)tile_h & 0xffffu) << 16);        // tensor_dim1[15:0]
    g1[3] = (int)(((unsigned)tile_w & 0xffffu) << 16);        // tile_dim0
    g1[4] = (int)((unsigned)tile_h & 0xffffu);                // tile_dim1 (tile_dim2=0)
    g1[5] = pitch;                                            // tensor_dim0_stride lo
    g1[6] = 0;
    g1[7] = 0;
    v4i z = {0, 0, 0, 0};
#if defined(__clang_major__) && (__clang_major__ >= 23)
    v8i z8 = {0, 0, 0, 0, 0, 0, 0, 0};
    __builtin_amdgcn_tensor_load_to_lds(g0, g1, z, z, z8, 0);
#else
    __builtin_amdgcn_tensor_load_to_lds(g0, g1, z, z, 0);
#endif
}
#endif // HAVE_TDM

// ---------------------------------------------------------------------------
// Generic WMMA GEMM:  C[M,N] = op( A[M,K](bf16) * Wp[K,N](bf16) + bias[N] )
// Wp is K-PAIR-INTERLEAVED: element (k,n) at Wp[(k/2)*2N + n*2 + (k&1)], so a
// B-fragment dword pair (B[k][n],B[k+1][n]) is one aligned u32 in LDS.
// Block tile 128x64, K step 32, 256 threads = 8 waves (wave32), each wave a
// 32x32 tile via 2x2 v_wmma_f32_16x16x32_bf16. Tiles staged by the Tensor
// Data Mover, double-buffered so the DMA of step k+32 overlaps step k's math.
// op: 0=identity, 1=silu, 2=softplus.
// ---------------------------------------------------------------------------
__global__ __launch_bounds__(256)
void gemm_bf16_kernel(const bf16_t* __restrict__ A, const bf16_t* __restrict__ W,
                      const float* __restrict__ bias,
                      float* __restrict__ Cf, bf16_t* __restrict__ Cb,
                      int M, int N, int K, int op)
{
    __shared__ __align__(16) bf16_t lA[2][128 * 32];
    __shared__ __align__(16) bf16_t lB[2][16 * 128];   // 16 pair-rows x (64 n x 2)

    const int tid  = threadIdx.x;
    const int lane = tid & 31;
    const int wid  = tid >> 5;
    const int wm   = wid & 3;    // 4 waves along M -> 32 rows each
    const int wn   = wid >> 2;   // 2 waves along N -> 32 cols each
    const int rowBase = blockIdx.y * 128;
    const int colBase = blockIdx.x * 64;
    const int r  = lane & 15;
    const int hi = lane >> 4;

    const bf16_t* Abase = A + (size_t)rowBase * K;
    const bf16_t* Wp    = W + (size_t)colBase * 2;     // pair-interleaved columns

    v8f acc[2][2] = {};

#if HAVE_TDM
    if (wid == 0) {
        tdm_load_2d((unsigned)(uintptr_t)&lA[0][0], Abase, 32, 128, K);
        tdm_load_2d((unsigned)(uintptr_t)&lB[0][0], Wp, 128, 16, 2 * N);
        __builtin_amdgcn_s_wait_tensorcnt(0);
    }
    __syncthreads();
#endif

    int buf = 0;
    for (int k0 = 0; k0 < K; k0 += 32) {
#if HAVE_TDM
        // async-stage the NEXT K tile while this one is consumed
        if (wid == 0 && k0 + 32 < K) {
            tdm_load_2d((unsigned)(uintptr_t)&lA[buf ^ 1][0], Abase + (k0 + 32), 32, 128, K);
            tdm_load_2d((unsigned)(uintptr_t)&lB[buf ^ 1][0],
                        Wp + (size_t)((k0 + 32) >> 1) * (2 * N), 128, 16, 2 * N);
        }
#else
#pragma unroll
        for (int it = 0; it < 2; ++it) {
            int c  = tid + it * 256;
            int ar = c >> 2;
            int ac = (c & 3) * 8;
            *(uint4*)&lA[buf][ar * 32 + ac] =
                *(const uint4*)&A[(size_t)(rowBase + ar) * K + k0 + ac];
        }
        {   // 16 pair-rows x 128 elements = 256 x 16B chunks
            int br = tid >> 4;
            int bc = (tid & 15) * 8;
            *(uint4*)&lB[buf][br * 128 + bc] =
                *(const uint4*)&Wp[(size_t)((k0 >> 1) + br) * (2 * N) + bc];
        }
        __syncthreads();
#endif

        // ---- fragment loads from LDS (contiguous k = 16*hi + s in both A and B) ----
        const unsigned int* lAu = (const unsigned int*)&lA[buf][0];
        const unsigned int* lBu = (const unsigned int*)&lB[buf][0];
        v16bf af[2], bfg[2];
#pragma unroll
        for (int i = 0; i < 2; ++i) {
            v8u t = {0, 0, 0, 0, 0, 0, 0, 0};
            int arow = wm * 32 + i * 16 + r;
#pragma unroll
            for (int g = 0; g < 8; ++g) {
                int kp = 2 * g + 16 * hi; // even
                t[g] = lAu[(arow * 32 + kp) >> 1];
            }
            af[i] = __builtin_bit_cast(v16bf, t);
        }
#pragma unroll
        for (int j = 0; j < 2; ++j) {
            v8u t = {0, 0, 0, 0, 0, 0, 0, 0};
            int bcol = wn * 32 + j * 16 + r;
#pragma unroll
            for (int g = 0; g < 8; ++g)               // pair (2g+16hi, 2g+1+16hi)
                t[g] = lBu[(g + 8 * hi) * 64 + bcol];
            bfg[j] = __builtin_bit_cast(v16bf, t);
        }

#pragma unroll
        for (int i = 0; i < 2; ++i)
#pragma unroll
            for (int j = 0; j < 2; ++j)
                acc[i][j] = __builtin_amdgcn_wmma_f32_16x16x32_bf16(
                    false, af[i], false, bfg[j], (short)0, acc[i][j], false, false);

#if HAVE_TDM
        if (wid == 0 && k0 + 32 < K) __builtin_amdgcn_s_wait_tensorcnt(0);
#endif
        __syncthreads();
        buf ^= 1;
    }

    // ---- epilogue: bias + activation + f32/bf16 stores ----
#pragma unroll
    for (int i = 0; i < 2; ++i) {
        int m0 = rowBase + wm * 32 + i * 16 + hi * 8;
#pragma unroll
        for (int j = 0; j < 2; ++j) {
            int n0 = colBase + wn * 32 + j * 16 + r;
            float bsv = bias ? bias[n0] : 0.f;
#pragma unroll
            for (int v = 0; v < 8; ++v) {
                float x = acc[i][j][v] + bsv;
                if (op == 1)      x = x / (1.f + __expf(-x));                 // silu
                else if (op == 2) x = (x > 20.f) ? x : log1pf(__expf(x));     // softplus
                size_t idx = (size_t)(m0 + v) * N + n0;
                if (Cf) Cf[idx] = x;
                if (Cb) Cb[idx] = (bf16_t)x;
            }
        }
    }
}

// ---------------------------------------------------------------------------
// Channel-mixing conv1d as 3 shift-accumulated WMMA GEMMs per batch:
//   XC[b,fo,l] = silu( sum_k sum_fi Wk[k][fo][fi] * XP[b,fi,l+k-1] + bias[fo] )
// Wk pre-repacked to [3][512][512]. The A (weight) tile is staged by TDM,
// overlapping the guarded manual staging of the shifted/zero-padded B tile
// (which cannot be expressed as a TDM descriptor at the left edge).
// ---------------------------------------------------------------------------
__global__ __launch_bounds__(256)
void conv_gemm_kernel(const bf16_t* __restrict__ Wk, const bf16_t* __restrict__ XP,
                      const float* __restrict__ bias, bf16_t* __restrict__ XC)
{
    __shared__ __align__(16) bf16_t lA[128 * 32];
    __shared__ __align__(16) bf16_t lB[32 * 64];

    const int tid  = threadIdx.x;
    const int lane = tid & 31;
    const int wid  = tid >> 5;
    const int wm   = wid & 3;
    const int wn   = wid >> 2;
    const int rowBase = blockIdx.y * 128;     // fo
    const int colBase = blockIdx.x * 64;      // l
    const int b       = blockIdx.z;
    const int r  = lane & 15;
    const int hi = lane >> 4;

    const bf16_t* Xb = XP + (size_t)b * 512 * 512;
    v8f acc[2][2] = {};
    const unsigned int* lAu = (const unsigned int*)lA;

    for (int kk = 0; kk < 3; ++kk) {
        const bf16_t* Ak = Wk + (size_t)kk * 512 * 512;
        const int s = kk - 1;                 // column shift (padding=1)
        for (int k0 = 0; k0 < 512; k0 += 32) {
#if HAVE_TDM
            if (wid == 0)                     // DMA weight tile; overlaps B staging
                tdm_load_2d((unsigned)(uintptr_t)lA,
                            Ak + (size_t)rowBase * 512 + k0, 32, 128, 512);
#else
#pragma unroll
            for (int it = 0; it < 2; ++it) {
                int c  = tid + it * 256;
                int ar = c >> 2;
                int ac = (c & 3) * 8;
                *(uint4*)&lA[ar * 32 + ac] =
                    *(const uint4*)&Ak[(size_t)(rowBase + ar) * 512 + k0 + ac];
            }
#endif
            {   // guarded (shifted) B tile loads
                int br  = tid >> 3;
                int bc0 = (tid & 7) * 8;
#pragma unroll
                for (int e = 0; e < 8; ++e) {
                    int l = colBase + bc0 + e + s;
                    bf16_t v = (l >= 0 && l < 512)
                                   ? Xb[(size_t)(k0 + br) * 512 + l]
                                   : (bf16_t)0.f;
                    lB[br * 64 + bc0 + e] = v;
                }
            }
#if HAVE_TDM
            if (wid == 0) __builtin_amdgcn_s_wait_tensorcnt(0);
#endif
            __syncthreads();

            v16bf af[2], bfg[2];
#pragma unroll
            for (int i = 0; i < 2; ++i) {
                v8u t = {0, 0, 0, 0, 0, 0, 0, 0};
                int arow = wm * 32 + i * 16 + r;
#pragma unroll
                for (int g = 0; g < 8; ++g) {
                    int kp = 2 * g + 16 * hi;
                    t[g] = lAu[(arow * 32 + kp) >> 1];
                }
                af[i] = __builtin_bit_cast(v16bf, t);
            }
#pragma unroll
            for (int j = 0; j < 2; ++j) {
                v8u t = {0, 0, 0, 0, 0, 0, 0, 0};
                int bcol = wn * 32 + j * 16 + r;
#pragma unroll
                for (int g = 0; g < 8; ++g) {
                    int k2 = 2 * g + 16 * hi;
                    unsigned lo = *(const unsigned short*)&lB[k2 * 64 + bcol];
                    unsigned h2 = *(const unsigned short*)&lB[(k2 + 1) * 64 + bcol];
                    t[g] = lo | (h2 << 16);
                }
                bfg[j] = __builtin_bit_cast(v16bf, t);
            }
#pragma unroll
            for (int i = 0; i < 2; ++i)
#pragma unroll
                for (int j = 0; j < 2; ++j)
                    acc[i][j] = __builtin_amdgcn_wmma_f32_16x16x32_bf16(
                        false, af[i], false, bfg[j], (short)0, acc[i][j], false, false);
            __syncthreads();
        }
    }

#pragma unroll
    for (int i = 0; i < 2; ++i) {
        int m0 = rowBase + wm * 32 + i * 16 + hi * 8;
#pragma unroll
        for (int j = 0; j < 2; ++j) {
            int n0 = colBase + wn * 32 + j * 16 + r;
#pragma unroll
            for (int v = 0; v < 8; ++v) {
                float x = acc[i][j][v] + bias[m0 + v];   // conv bias is per fo (row)
                x = x / (1.f + __expf(-x));              // silu
                XC[((size_t)b * 512 + m0 + v) * 512 + n0] = (bf16_t)x;
            }
        }
    }
}

// ---------------------------------------------------------------------------
// Small kernels: normalization, reductions, elementwise, precision packing.
// ---------------------------------------------------------------------------

__global__ __launch_bounds__(256)
void norm_concat_kernel(const float* __restrict__ xe, const float* __restrict__ xm,
                        float* __restrict__ X, float* __restrict__ means,
                        float* __restrict__ stds)
{
    __shared__ float sm[256];
    const int bc = blockIdx.x;            // b*512 + c
    const int b  = bc >> 9, c = bc & 511;
    const int t  = threadIdx.x;
    if (c < 508) {
        float v = xe[((size_t)b * 256 + t) * 508 + c];
        sm[t] = v; __syncthreads();
        for (int s = 128; s > 0; s >>= 1) { if (t < s) sm[t] += sm[t + s]; __syncthreads(); }
        float mean = sm[0] * (1.f / 256.f);
        __syncthreads();
        float d = v - mean;
        sm[t] = d * d; __syncthreads();
        for (int s = 128; s > 0; s >>= 1) { if (t < s) sm[t] += sm[t + s]; __syncthreads(); }
        float sd = sqrtf(sm[0] * (1.f / 256.f) + 1e-5f);
        X[(size_t)bc * 256 + t] = d / sd;
        if (t == 0) { means[bc] = mean; stds[bc] = sd; }
    } else {
        X[(size_t)bc * 256 + t] = xm[((size_t)b * 256 + t) * 4 + (c - 508)];
        if (t == 0) { means[bc] = 0.f; stds[bc] = 1.f; }
    }
}

__global__ __launch_bounds__(256)
void rmsnorm_kernel(const float* __restrict__ X, const float* __restrict__ w,
                    bf16_t* __restrict__ Y)
{
    __shared__ float sm[256];
    const int row = blockIdx.x, t = threadIdx.x;
    float x = X[(size_t)row * 256 + t];
    sm[t] = x * x; __syncthreads();
    for (int s = 128; s > 0; s >>= 1) { if (t < s) sm[t] += sm[t + s]; __syncthreads(); }
    float r = rsqrtf(sm[0] * (1.f / 256.f) + 1e-5f);
    Y[(size_t)row * 256 + t] = (bf16_t)(x * r * w[t]);
}

__global__ __launch_bounds__(256)
void cb_kernel(const float* __restrict__ Bm, const float* __restrict__ Cm,
               float* __restrict__ cb)
{
    __shared__ float sm[256];
    const int row = blockIdx.x, t = threadIdx.x;
    size_t i = (size_t)row * 256 + t;
    sm[t] = Bm[i] * Cm[i]; __syncthreads();
    for (int s = 128; s > 0; s >>= 1) { if (t < s) sm[t] += sm[t + s]; __syncthreads(); }
    if (t == 0) cb[row] = sm[0];
}

__global__ __launch_bounds__(256)
void ssm_mul_kernel(const float* __restrict__ delta, const float* __restrict__ xco,
                    const float* __restrict__ cb, const float* __restrict__ xres,
                    bf16_t* __restrict__ Z)
{
    size_t idx = (size_t)blockIdx.x * 256 + threadIdx.x;
    size_t row = idx >> 9;   // cols = 512
    float x = delta[idx] * xco[idx] * cb[row];
    x = x / (1.f + __expf(-x));
    Z[idx] = (bf16_t)(x * xres[idx]);
}

__global__ __launch_bounds__(256)
void final_kernel(const float* __restrict__ P, const float* __restrict__ means,
                  const float* __restrict__ stds, float* __restrict__ out, size_t total)
{
    size_t idx = (size_t)blockIdx.x * 256 + threadIdx.x;
    if (idx >= total) return;
    int n = (int)(idx % 508);
    int t = (int)((idx / 508) % 256);
    int b = (int)(idx / (508 * 256));
    size_t bc = (size_t)b * 512 + n;
    out[idx] = P[bc * 256 + t] * stds[bc] + means[bc];
}

// f32 [batch][K][N] -> bf16 K-pair-interleaved: dst[(k/2)*2N + n*2 + (k&1)]
__global__ void pack_w_kernel(const float* __restrict__ src, bf16_t* __restrict__ dst,
                              int K, int N, int total)
{
    int i = blockIdx.x * 256 + threadIdx.x;
    if (i >= total) return;
    int n  = i % N;
    int k  = (i / N) % K;
    int bt = i / (K * N);
    dst[(size_t)bt * K * N + (size_t)(k >> 1) * 2 * N + n * 2 + (k & 1)] = (bf16_t)src[i];
}

// conv_W [3][fo][fi][k] (f32) -> [3][k][fo][fi] (bf16), GEMM-friendly (A side)
__global__ void conv_repack_kernel(const float* __restrict__ W, bf16_t* __restrict__ out)
{
    int i = blockIdx.x * 256 + threadIdx.x;
    if (i >= 3 * 3 * 512 * 512) return;
    int fi  = i & 511;
    int fo  = (i >> 9) & 511;
    int k   = (i / (512 * 512)) % 3;
    int blk = i / (3 * 512 * 512);
    out[i] = (bf16_t)W[(((size_t)blk * 512 + fo) * 512 + fi) * 3 + k];
}

// ---------------------------------------------------------------------------
// Host orchestration
// ---------------------------------------------------------------------------
extern "C" void kernel_launch(void* const* d_in, const int* in_sizes, int n_in,
                              void* d_out, int out_size, void* d_ws, size_t ws_size,
                              hipStream_t stream)
{
    (void)in_sizes; (void)n_in; (void)out_size; (void)ws_size;
    const float* x_enc     = (const float*)d_in[0];
    const float* x_mark    = (const float*)d_in[1];
    const float* norm_w    = (const float*)d_in[4];
    const float* inp_W     = (const float*)d_in[5];
    const float* inp_b     = (const float*)d_in[6];
    const float* conv_W    = (const float*)d_in[7];
    const float* conv_b    = (const float*)d_in[8];
    const float* convlin_W = (const float*)d_in[9];
    const float* convlin_b = (const float*)d_in[10];
    const float* fc1_W     = (const float*)d_in[11];
    const float* fc1_b     = (const float*)d_in[12];
    const float* fc2_W     = (const float*)d_in[13];
    const float* fc2_b     = (const float*)d_in[14];
    const float* fc3_W     = (const float*)d_in[15];
    const float* fc3_b     = (const float*)d_in[16];
    const float* D_W       = (const float*)d_in[17];
    const float* D_b       = (const float*)d_in[18];
    const float* out_W     = (const float*)d_in[19];
    const float* out_b     = (const float*)d_in[20];
    const float* proj_W    = (const float*)d_in[21];
    const float* proj_b    = (const float*)d_in[22];
    float* out = (float*)d_out;

    const int ROWS = 32 * 512;   // 16384
    char* ws = (char*)d_ws;
    size_t off = 0;
    auto alloc = [&](size_t bytes) -> void* {
        off = (off + 255) & ~(size_t)255;
        void* p = ws + off;
        off += bytes;
        return p;
    };

    float*  wX     = (float*) alloc((size_t)ROWS * 256 * 4);
    float*  wP     = (float*) alloc((size_t)ROWS * 256 * 4);
    float*  wMeans = (float*) alloc((size_t)ROWS * 4);
    float*  wStd   = (float*) alloc((size_t)ROWS * 4);
    float*  wXCOf  = (float*) alloc((size_t)ROWS * 512 * 4);
    float*  wDelta = (float*) alloc((size_t)ROWS * 512 * 4);
    float*  wBm    = (float*) alloc((size_t)ROWS * 256 * 4);
    float*  wCm    = (float*) alloc((size_t)ROWS * 256 * 4);
    float*  wCb    = (float*) alloc((size_t)ROWS * 4);
    float*  wXres  = (float*) alloc((size_t)ROWS * 512 * 4);
    bf16_t* wXb    = (bf16_t*)alloc((size_t)ROWS * 256 * 2);
    bf16_t* wXn    = (bf16_t*)alloc((size_t)ROWS * 256 * 2);
    bf16_t* wXP    = (bf16_t*)alloc((size_t)ROWS * 512 * 2);
    bf16_t* wXC    = (bf16_t*)alloc((size_t)ROWS * 512 * 2);
    bf16_t* wXCOb  = (bf16_t*)alloc((size_t)ROWS * 512 * 2);
    bf16_t* wZ     = (bf16_t*)alloc((size_t)ROWS * 512 * 2);
    bf16_t* inpWb     = (bf16_t*)alloc((size_t)3 * 256 * 512 * 2);
    bf16_t* convWkb   = (bf16_t*)alloc((size_t)3 * 3 * 512 * 512 * 2);
    bf16_t* convlinWb = (bf16_t*)alloc((size_t)3 * 512 * 512 * 2);
    bf16_t* fc1Wb     = (bf16_t*)alloc((size_t)3 * 512 * 512 * 2);
    bf16_t* fc2Wb     = (bf16_t*)alloc((size_t)3 * 512 * 256 * 2);
    bf16_t* fc3Wb     = (bf16_t*)alloc((size_t)3 * 512 * 256 * 2);
    bf16_t* DWb       = (bf16_t*)alloc((size_t)3 * 256 * 512 * 2);
    bf16_t* outWb     = (bf16_t*)alloc((size_t)3 * 512 * 256 * 2);
    bf16_t* projWb    = (bf16_t*)alloc((size_t)256 * 256 * 2);

    auto pack = [&](const float* s, bf16_t* d, int K, int N, int batch) {
        int total = batch * K * N;
        pack_w_kernel<<<(total + 255) / 256, 256, 0, stream>>>(s, d, K, N, total);
    };
    pack(inp_W,     inpWb,     256, 512, 3);
    pack(convlin_W, convlinWb, 512, 512, 3);
    pack(fc1_W,     fc1Wb,     512, 512, 3);
    pack(fc2_W,     fc2Wb,     512, 256, 3);
    pack(fc3_W,     fc3Wb,     512, 256, 3);
    pack(D_W,       DWb,       256, 512, 3);
    pack(out_W,     outWb,     512, 256, 3);
    pack(proj_W,    projWb,    256, 256, 1);
    conv_repack_kernel<<<(3 * 3 * 512 * 512 + 255) / 256, 256, 0, stream>>>(conv_W, convWkb);

    auto gemm = [&](const bf16_t* A, const bf16_t* W, const float* bias,
                    float* Cf, bf16_t* Cb, int M, int N, int K, int op) {
        dim3 g(N / 64, M / 128);
        gemm_bf16_kernel<<<g, 256, 0, stream>>>(A, W, bias, Cf, Cb, M, N, K, op);
    };

    norm_concat_kernel<<<ROWS, 256, 0, stream>>>(x_enc, x_mark, wX, wMeans, wStd);

    for (int i = 0; i < 3; ++i) {
        rmsnorm_kernel<<<ROWS, 256, 0, stream>>>(wX, norm_w + i * 256, wXn);
        gemm(wXn, inpWb + (size_t)i * 256 * 512, inp_b + i * 512,
             nullptr, wXP, ROWS, 512, 256, 0);
        conv_gemm_kernel<<<dim3(8, 4, 32), 256, 0, stream>>>(
            convWkb + (size_t)i * 3 * 512 * 512, wXP, conv_b + i * 512, wXC);
        gemm(wXC, convlinWb + (size_t)i * 512 * 512, convlin_b + i * 512,
             wXCOf, wXCOb, ROWS, 512, 512, 0);
        gemm(wXCOb, fc1Wb + (size_t)i * 512 * 512, fc1_b + i * 512,
             wDelta, nullptr, ROWS, 512, 512, 2);                       // softplus
        gemm(wXCOb, fc2Wb + (size_t)i * 512 * 256, fc2_b + i * 256,
             wBm, nullptr, ROWS, 256, 512, 0);
        gemm(wXCOb, fc3Wb + (size_t)i * 512 * 256, fc3_b + i * 256,
             wCm, nullptr, ROWS, 256, 512, 0);
        cb_kernel<<<ROWS, 256, 0, stream>>>(wBm, wCm, wCb);
        gemm(wXn, DWb + (size_t)i * 256 * 512, D_b + i * 512,
             wXres, nullptr, ROWS, 512, 256, 1);                        // silu
        ssm_mul_kernel<<<ROWS * 2, 256, 0, stream>>>(wDelta, wXCOf, wCb, wXres, wZ);
        gemm(wZ, outWb + (size_t)i * 512 * 256, out_b + i * 256,
             wX, wXb, ROWS, 256, 512, 0);                               // wX = next block input
    }

    gemm(wXb, projWb, proj_b, wP, nullptr, ROWS, 256, 256, 0);
    size_t total = (size_t)32 * 256 * 508;
    final_kernel<<<(unsigned)((total + 255) / 256), 256, 0, stream>>>(wP, wMeans, wStd, out, total);
}
